// MultiheadDiffAttn_64201171140828
// MI455X (gfx1250) — matrix operand order repaired and checked by reading, not compiled, not run on hardware
//
#include <hip/hip_runtime.h>
#include <hip/hip_bf16.h>

// ---------------------------------------------------------------------------
// MultiheadDiffAttn for MI455X (gfx1250, wave32, WMMA bf16 path)
//   B=1, T=2048, E=1024, hd=32, H=16 heads (2H=32 half-heads), v-dim 64.
// Pipeline:
//   (1) f32->bf16 convert x, Wq, Wk, Wv, Wo
//   (2) WMMA GEMM (64x64 per wave):  Q = x@Wq^T, K = x@Wk^T, V^T = (x@Wv^T)^T
//   (3) flash-style diff-attention per (head, 16-query tile) wave:
//       double-buffered async global->LDS staging of K/V tiles,
//       online two-path softmax, P staged via LDS to A-fragment layout,
//       PV via WMMA, fused lambda-diff + RMSNorm + gamma + (1-lambda_init)
//   (4) WMMA GEMM:  out = attn@Wo^T (f32)
// ---------------------------------------------------------------------------

#define T_SEQ 2048
#define E_DIM 1024
#define HD    32
#define NHEAD 16

#define LAMBDA_INIT_F     0.47071301834358421f   /* 0.8 - 0.6*exp(-0.6) */
#define ONE_MINUS_LI_F    0.52928698165641579f
#define QK_SCALE_F        0.17677669529663687f   /* 32^-0.5 */

typedef __attribute__((ext_vector_type(16))) __bf16 v16bf;
typedef __attribute__((ext_vector_type(8)))  __bf16 v8bf;
typedef __attribute__((ext_vector_type(8)))  float  v8f;

// ---- CDNA5 async global->LDS path (guarded; falls back to direct loads) ---
#if defined(__gfx1250__) && __has_builtin(__builtin_amdgcn_global_load_async_to_lds_b128)
#define HAS_ASYNC_G2L 1
typedef int v4i __attribute__((vector_size(16)));
typedef __attribute__((address_space(1))) v4i gv4i_t;   // global 16B vector
typedef __attribute__((address_space(3))) v4i lv4i_t;   // LDS 16B vector
static __device__ __forceinline__ void g2l_b128(const unsigned short* g,
                                                unsigned short* l) {
  // generic global pointer value == AS1 value; low 32 bits of a generic LDS
  // pointer are the AS3 offset (ISA flat-addressing rule).
  __builtin_amdgcn_global_load_async_to_lds_b128(
      (gv4i_t*)(unsigned long long)g,
      (lv4i_t*)(unsigned)(unsigned long long)l, 0, 0);
}
#if __has_builtin(__builtin_amdgcn_s_wait_asynccnt)
#define WAIT_ASYNC(n) __builtin_amdgcn_s_wait_asynccnt(n)
#else
#define WAIT_ASYNC(n) asm volatile("s_wait_asynccnt %0" ::"i"(n) : "memory")
#endif
#else
#define HAS_ASYNC_G2L 0
#endif

static __device__ __forceinline__ unsigned short f32_to_bf16(float f) {
  union { float f; unsigned u; } v; v.f = f;
  unsigned r = v.u + 0x7FFFu + ((v.u >> 16) & 1u);   // round-to-nearest-even
  return (unsigned short)(r >> 16);
}

// A-fragment (16x32 bf16): lane lo = M, elements e:
//   K = e + hi*16  -> one contiguous 16-elem run (two 8-elem loads).
static __device__ __forceinline__ v16bf load_a_frag(const unsigned short* p) {
  v8bf x0 = *(const v8bf*)(p);
  v8bf x1 = *(const v8bf*)(p + 8);
  v16bf r;
#pragma unroll
  for (int i = 0; i < 8; ++i) { r[i] = x0[i]; r[i + 8] = x1[i]; }
  return r;
}

static __device__ __forceinline__ v8f wmma_bf16(v16bf a, v16bf b, v8f c) {
  return __builtin_amdgcn_wmma_f32_16x16x32_bf16(false, a, false, b,
                                                 (short)0, c, false, false);
}

static __device__ __forceinline__ float rmax16(float v) {
#pragma unroll
  for (int m = 8; m >= 1; m >>= 1) v = fmaxf(v, __shfl_xor(v, m, 32));
  return v;
}
static __device__ __forceinline__ float rsum16(float v) {
#pragma unroll
  for (int m = 8; m >= 1; m >>= 1) v += __shfl_xor(v, m, 32);
  return v;
}
static __device__ __forceinline__ float rsum32(float v) {
#pragma unroll
  for (int m = 16; m >= 1; m >>= 1) v += __shfl_xor(v, m, 32);
  return v;
}

// ---------------------------------------------------------------------------
__global__ __launch_bounds__(256) void cvt_f32_bf16_kernel(
    const float* __restrict__ src, unsigned short* __restrict__ dst, int n) {
  int i = blockIdx.x * blockDim.x + threadIdx.x;
  if (i < n) dst[i] = f32_to_bf16(src[i]);
}

// ---------------------------------------------------------------------------
// GEMM: Out[M,N] = A[M,K] * W[N,K]^T, bf16 inputs, one wave -> 64x64 tile
// (16 WMMAs per 32-wide K step; ~16B loaded per lane per WMMA).
// MODE 0: bf16 row-major; MODE 1: bf16 transposed (Out[n*M+m]); MODE 2: f32.
template <int MODE>
__global__ __launch_bounds__(32) void gemm_bf16_wmma(
    const unsigned short* __restrict__ A, const unsigned short* __restrict__ W,
    void* __restrict__ Out, int M, int N, int K) {
  const int lane = threadIdx.x;
  const int lo = lane & 15, hi = lane >> 4;
  const int n0 = blockIdx.x * 64;
  const int m0 = blockIdx.y * 64;

  v8f c[4][4] = {};

  for (int k0 = 0; k0 < K; k0 += 32) {
    v16bf a[4];
#pragma unroll
    for (int mt = 0; mt < 4; ++mt)
      a[mt] = load_a_frag(A + (size_t)(m0 + mt * 16 + lo) * K + k0 + hi * 16);
#pragma unroll
    for (int nt = 0; nt < 4; ++nt) {
      // B-fragment (32x16): lane lo = N column, elements = contiguous K.
      v16bf b = *(const v16bf*)(W + (size_t)(n0 + nt * 16 + lo) * K + k0 + hi * 16);
#pragma unroll
      for (int mt = 0; mt < 4; ++mt) c[mt][nt] = wmma_bf16(a[mt], b, c[mt][nt]);
    }
  }

  // C layout: VGPR r -> (M = r + 8*hi, N = lane&15)
#pragma unroll
  for (int mt = 0; mt < 4; ++mt) {
#pragma unroll
    for (int nt = 0; nt < 4; ++nt) {
#pragma unroll
      for (int r = 0; r < 8; ++r) {
        const int m = m0 + mt * 16 + r + 8 * hi;
        const int n = n0 + nt * 16 + lo;
        if (MODE == 0) {
          ((unsigned short*)Out)[(size_t)m * N + n] = f32_to_bf16(c[mt][nt][r]);
        } else if (MODE == 1) {
          ((unsigned short*)Out)[(size_t)n * M + m] = f32_to_bf16(c[mt][nt][r]);
        } else {
          ((float*)Out)[(size_t)m * N + n] = c[mt][nt][r];
        }
      }
    }
  }
}

// ---------------------------------------------------------------------------
// Flash-style differential attention.  One wave = (head h, 16 query rows).
__global__ __launch_bounds__(32) void diffattn_wmma_kernel(
    const unsigned short* __restrict__ Qbf, const unsigned short* __restrict__ Kbf,
    const unsigned short* __restrict__ vT,  const float* __restrict__ gamma,
    const float* __restrict__ lq1, const float* __restrict__ lk1,
    const float* __restrict__ lq2, const float* __restrict__ lk2,
    unsigned short* __restrict__ attnbf) {
  __shared__ __align__(16) unsigned short sP0[16 * 32];
  __shared__ __align__(16) unsigned short sP1[16 * 32];
#if HAS_ASYNC_G2L
  __shared__ __align__(16) unsigned short sK[2][32 * 64];  // [key][64 dims]
  __shared__ __align__(16) unsigned short sV[2][64 * 32];  // [col][32 keys]
#endif

  const int lane = threadIdx.x;
  const int lo = lane & 15, hi = lane >> 4;
  const int h  = blockIdx.y;
  const int t0 = blockIdx.x * 16;
  const int tmax = t0 + 15;

  // lambda_full = exp(lq1.lk1) - exp(lq2.lk2) + lambda_init  (hd==warpSize==32)
  float d1 = lq1[lane] * lk1[lane];
  float d2 = lq2[lane] * lk2[lane];
  d1 = rsum32(d1);  d2 = rsum32(d2);
  const float lambda_full = __expf(d1) - __expf(d2) + LAMBDA_INIT_F;

  const int c0 = (2 * h) * HD;       // half-head 0 column base (half-head 1 is +32,
                                     // contiguous: the head uses 64 consecutive cols)

  // Q fragments, loaded once (A layout).
  const unsigned short* qrow = Qbf + (size_t)(t0 + lo) * E_DIM;
  const v16bf q0 = load_a_frag(qrow + c0 + hi * 16);
  const v16bf q1 = load_a_frag(qrow + c0 + 32 + hi * 16);

  v8f acc0[4] = {}, acc1[4] = {};
  float m0v[8], l0v[8], m1v[8], l1v[8];
#pragma unroll
  for (int r = 0; r < 8; ++r) { m0v[r] = m1v[r] = -1e30f; l0v[r] = l1v[r] = 0.f; }

  const int nsteps = tmax / 32 + 1;       // causal: keys 0..tmax, 32 per step

#if HAS_ASYNC_G2L
  // Async-stage one 32-key step: K tile 32x(32dims) = 2KB (4 b128 ops),
  // V^T tile 64x(32keys) = 4KB (8 b128 ops).  12 ASYNCcnt increments/step.
  auto issue_step = [&](int step, int buf) {
#pragma unroll
    for (int i = 0; i < 4; ++i) {                    // K tile
      const int id = i * 32 + lane, s = id >> 2, j = id & 3;
      g2l_b128(Kbf + (size_t)(step * 32 + s) * E_DIM + c0 + j * 8,
               &sK[buf][s * 64 + j * 8]);
    }
#pragma unroll
    for (int i = 0; i < 8; ++i) {                    // V^T tile
      const int id = i * 32 + lane, c = id >> 2, j = id & 3;
      g2l_b128(vT + (size_t)(h * 64 + c) * T_SEQ + step * 32 + j * 8,
               &sV[buf][c * 32 + j * 8]);
    }
  };
  issue_step(0, 0);
#endif

  for (int step = 0; step < nsteps; ++step) {
    const int s0 = step * 32;
    const int buf = step & 1;
#if HAS_ASYNC_G2L
    if (step + 1 < nsteps) {
      issue_step(step + 1, buf ^ 1);   // prefetch next step into other buffer
      WAIT_ASYNC(12);                  // in-order: current buffer has landed
    } else {
      WAIT_ASYNC(0);
    }
    __syncthreads();
#endif

    // ---- S = (Q K^T) * scale, causal mask ------------------------------
    v8f S0[2], S1[2];
#pragma unroll
    for (int ct = 0; ct < 2; ++ct) {
#if HAS_ASYNC_G2L
      v16bf kb0 = *(const v16bf*)&sK[buf][(ct * 16 + lo) * 64 + hi * 16];
      v16bf kb1 = *(const v16bf*)&sK[buf][(ct * 16 + lo) * 64 + 32 + hi * 16];
#else
      const unsigned short* krow = Kbf + (size_t)(s0 + ct * 16 + lo) * E_DIM;
      v16bf kb0 = *(const v16bf*)(krow + c0 + hi * 16);
      v16bf kb1 = *(const v16bf*)(krow + c0 + 32 + hi * 16);
#endif
      v8f z = {};
      S0[ct] = wmma_bf16(q0, kb0, z);
      S1[ct] = wmma_bf16(q1, kb1, z);
    }
#pragma unroll
    for (int ct = 0; ct < 2; ++ct) {
#pragma unroll
      for (int r = 0; r < 8; ++r) {
        const int s = s0 + ct * 16 + lo;
        const int t = t0 + r + 8 * hi;
        float v0 = S0[ct][r] * QK_SCALE_F;
        float v1 = S1[ct][r] * QK_SCALE_F;
        if (s > t) { v0 = -1e30f; v1 = -1e30f; }
        S0[ct][r] = v0; S1[ct][r] = v1;
      }
    }

    // ---- online softmax (both paths), P -> LDS as bf16 -----------------
#pragma unroll
    for (int r = 0; r < 8; ++r) {
      const int row = r + 8 * hi;
      // path 0
      {
        float mx = rmax16(fmaxf(S0[0][r], S0[1][r]));
        float mn = fmaxf(m0v[r], mx);
        float alpha = __expf(m0v[r] - mn);
        float pa = __expf(S0[0][r] - mn);
        float pb = __expf(S0[1][r] - mn);
        float rs = rsum16(pa + pb);
        l0v[r] = l0v[r] * alpha + rs;
        m0v[r] = mn;
#pragma unroll
        for (int nt = 0; nt < 4; ++nt) acc0[nt][r] *= alpha;
        sP0[row * 32 + lo]      = f32_to_bf16(pa);
        sP0[row * 32 + 16 + lo] = f32_to_bf16(pb);
      }
      // path 1
      {
        float mx = rmax16(fmaxf(S1[0][r], S1[1][r]));
        float mn = fmaxf(m1v[r], mx);
        float alpha = __expf(m1v[r] - mn);
        float pa = __expf(S1[0][r] - mn);
        float pb = __expf(S1[1][r] - mn);
        float rs = rsum16(pa + pb);
        l1v[r] = l1v[r] * alpha + rs;
        m1v[r] = mn;
#pragma unroll
        for (int nt = 0; nt < 4; ++nt) acc1[nt][r] *= alpha;
        sP1[row * 32 + lo]      = f32_to_bf16(pa);
        sP1[row * 32 + 16 + lo] = f32_to_bf16(pb);
      }
    }
    __syncthreads();

    // ---- PV: re-load P as A fragments ----------------------------------
    const v16bf pa0 = load_a_frag(&sP0[lo * 32 + hi * 16]);
    const v16bf pa1 = load_a_frag(&sP1[lo * 32 + hi * 16]);
#pragma unroll
    for (int nt = 0; nt < 4; ++nt) {
#if HAS_ASYNC_G2L
      v16bf vb = *(const v16bf*)&sV[buf][(nt * 16 + lo) * 32 + hi * 16];
#else
      v16bf vb = *(const v16bf*)(vT + (size_t)(h * 64 + nt * 16 + lo) * T_SEQ +
                                 s0 + hi * 16);
#endif
      acc0[nt] = wmma_bf16(pa0, vb, acc0[nt]);
      acc1[nt] = wmma_bf16(pa1, vb, acc1[nt]);
    }
    __syncthreads();
  }

  // ---- epilogue: diff, RMSNorm over 64, gamma, (1-lambda_init) ----------
  float g[4];
#pragma unroll
  for (int nt = 0; nt < 4; ++nt) g[nt] = gamma[nt * 16 + lo];

#pragma unroll
  for (int r = 0; r < 8; ++r) {
    const float inv0 = 1.0f / l0v[r];
    const float inv1 = lambda_full / l1v[r];
    float vals[4]; float ss = 0.f;
#pragma unroll
    for (int nt = 0; nt < 4; ++nt) {
      vals[nt] = acc0[nt][r] * inv0 - acc1[nt][r] * inv1;
      ss += vals[nt] * vals[nt];
    }
    ss = rsum16(ss);
    const float rn = rsqrtf(ss * (1.0f / 64.0f) + 1e-5f);
    const int t = t0 + r + 8 * hi;
#pragma unroll
    for (int nt = 0; nt < 4; ++nt) {
      attnbf[(size_t)t * E_DIM + h * 64 + nt * 16 + lo] =
          f32_to_bf16(g[nt] * vals[nt] * rn * ONE_MINUS_LI_F);
    }
  }
}

// ---------------------------------------------------------------------------
extern "C" void kernel_launch(void* const* d_in, const int* in_sizes, int n_in,
                              void* d_out, int out_size, void* d_ws, size_t ws_size,
                              hipStream_t stream) {
  (void)in_sizes; (void)n_in; (void)out_size;
  const float* x     = (const float*)d_in[0];
  const float* Wq    = (const float*)d_in[1];
  const float* Wk    = (const float*)d_in[2];
  const float* Wv    = (const float*)d_in[3];
  const float* Wo    = (const float*)d_in[4];
  const float* lq1   = (const float*)d_in[5];
  const float* lk1   = (const float*)d_in[6];
  const float* lq2   = (const float*)d_in[7];
  const float* lk2   = (const float*)d_in[8];
  const float* gamma = (const float*)d_in[9];

  const int NX = T_SEQ * E_DIM;   // 2,097,152 elems
  const int NW = E_DIM * E_DIM;   // 1,048,576 elems

  // Workspace (bf16/u16 elements), 28 MiB total.
  unsigned short* ws     = (unsigned short*)d_ws;
  unsigned short* xbf    = ws;
  unsigned short* Wqbf   = xbf  + NX;
  unsigned short* Wkbf   = Wqbf + NW;
  unsigned short* Wvbf   = Wkbf + NW;
  unsigned short* Wobf   = Wvbf + NW;
  unsigned short* Qbf    = Wobf + NW;
  unsigned short* Kbf    = Qbf  + NX;
  unsigned short* vTbf   = Kbf  + NX;   // [E_DIM][T_SEQ] = V transposed
  unsigned short* attnbf = vTbf + NX;
  (void)ws_size;

  // (1) convert to bf16
  cvt_f32_bf16_kernel<<<(NX + 255) / 256, 256, 0, stream>>>(x,  xbf,  NX);
  cvt_f32_bf16_kernel<<<(NW + 255) / 256, 256, 0, stream>>>(Wq, Wqbf, NW);
  cvt_f32_bf16_kernel<<<(NW + 255) / 256, 256, 0, stream>>>(Wk, Wkbf, NW);
  cvt_f32_bf16_kernel<<<(NW + 255) / 256, 256, 0, stream>>>(Wv, Wvbf, NW);
  cvt_f32_bf16_kernel<<<(NW + 255) / 256, 256, 0, stream>>>(Wo, Wobf, NW);

  // (2) projections: grid = (N/64, M/64) waves of 32
  const dim3 gGemm(E_DIM / 64, T_SEQ / 64);
  gemm_bf16_wmma<0><<<gGemm, 32, 0, stream>>>(xbf, Wqbf, Qbf,  T_SEQ, E_DIM, E_DIM);
  gemm_bf16_wmma<0><<<gGemm, 32, 0, stream>>>(xbf, Wkbf, Kbf,  T_SEQ, E_DIM, E_DIM);
  gemm_bf16_wmma<1><<<gGemm, 32, 0, stream>>>(xbf, Wvbf, vTbf, T_SEQ, E_DIM, E_DIM);

  // (3) differential attention: grid = (T/16 query tiles, H heads)
  diffattn_wmma_kernel<<<dim3(T_SEQ / 16, NHEAD), 32, 0, stream>>>(
      Qbf, Kbf, vTbf, gamma, lq1, lk1, lq2, lk2, attnbf);

  // (4) output projection -> f32
  gemm_bf16_wmma<2><<<gGemm, 32, 0, stream>>>(attnbf, Wobf, d_out,
                                              T_SEQ, E_DIM, E_DIM);
}
